// PAM1_module_34059090657715
// MI455X (gfx1250) — matrix operand order repaired and checked by reading, not compiled, not run on hardware
//
#include <hip/hip_runtime.h>
#include <hip/hip_bf16.h>

// ---------------------------------------------------------------------------
// PAM (position attention + relative position) for MI455X / gfx1250.
// Compute-bound (~60 GFLOP vs ~38 MB I/O after factoring V*(P*Q) = (V*P)*Q),
// so everything heavy runs through v_wmma_f32_16x16x32_f16.
// GEMM convention: C = A * B^T with A (MxK) and Bt (NxK) both row-major ->
// all LDS fragment reads are contiguous ds_load_b128 (no 16-bit gathers).
// Tiles are staged with gfx1250 async global->LDS copies, double-buffered so
// the next K-slab streams in while WMMAs run on the current one.
// ---------------------------------------------------------------------------

typedef __attribute__((ext_vector_type(16))) _Float16 v16h;
typedef __attribute__((ext_vector_type(8)))  _Float16 v8h;
typedef __attribute__((ext_vector_type(8)))  float    v8f;

#define kB   8
#define kC   512
#define kH   48
#define kW   48
#define kDK  64
#define kN   (kH * kW)      // 2304
#define kL   48
#define kR   (2 * kL - 1)   // 95

#define TILE_M 64
#define TILE_N 128
#define TILE_K 64
#define KPAD   8            // row padding in halves (keeps 16B alignment: 144B rows)

// 16-byte int vector, matching the async-LDS builtin's parameter type
typedef int i32x4 __attribute__((vector_size(16)));
typedef __attribute__((address_space(1))) i32x4 g_i32x4;   // global (__device__)
typedef __attribute__((address_space(3))) i32x4 l_i32x4;   // LDS (__shared__)

// 16-byte global->LDS copy: async (ASYNCcnt, bypasses VGPRs) when the gfx1250
// builtin exists, else plain load + ds_store. OOB lanes zero-fill LDS.
__device__ __forceinline__ void stage16(const _Float16* __restrict__ g,
                                        _Float16* l, bool valid)
{
#if __has_builtin(__builtin_amdgcn_global_load_async_to_lds_b128)
    if (valid) {
        g_i32x4* gp = (g_i32x4*)(unsigned long long)g;
        l_i32x4* lp = (l_i32x4*)(unsigned)(unsigned long long)l;
        __builtin_amdgcn_global_load_async_to_lds_b128(gp, lp, 0, 0);
    } else {
        *reinterpret_cast<uint4*>(l) = make_uint4(0u, 0u, 0u, 0u);
    }
#else
    uint4 v = valid ? *reinterpret_cast<const uint4*>(g) : make_uint4(0u, 0u, 0u, 0u);
    *reinterpret_cast<uint4*>(l) = v;
#endif
}

__device__ __forceinline__ void wait_async_stage()
{
#if __has_builtin(__builtin_amdgcn_s_wait_asynccnt)
    __builtin_amdgcn_s_wait_asynccnt(0);
#elif __has_builtin(__builtin_amdgcn_global_load_async_to_lds_b128)
    asm volatile("s_wait_asynccnt 0x0" ::: "memory");
#endif
}

// ---------------------------------------------------------------------------
// NT GEMM: C[b](MxN, f32) = A[b](MxK) * Bt[b](NxK)^T (+ bias[m]).
// 256 threads = 8 waves; block tile 64x128; wave tile 32x32 (2x2 WMMA frags);
// K staged in double-buffered slabs of 64 (all K here are multiples of 64).
// BOUNDED=false is used when M%64==0 && N%128==0 (every large GEMM in this
// pipeline): no exec-mask bounds juggling in staging, unguarded epilogue.
// ---------------------------------------------------------------------------
template <bool BOUNDED>
__global__ __launch_bounds__(256)
void wmma_gemm_nt(const _Float16* __restrict__ A, const _Float16* __restrict__ Bt,
                  float* __restrict__ Cc, const float* __restrict__ bias,
                  int M, int N, int K,
                  long long sA, long long sB, long long sC)
{
    __shared__ __align__(16) _Float16 As[2][TILE_M][TILE_K + KPAD];
    __shared__ __align__(16) _Float16 Bs[2][TILE_N][TILE_K + KPAD];

    const int bz = blockIdx.z;
    A  += (size_t)bz * sA;
    Bt += (size_t)bz * sB;
    Cc += (size_t)bz * sC;

    const int tid   = threadIdx.x;
    const int lane  = tid & 31;
    const int wave  = tid >> 5;
    const int waveM = wave >> 2;          // 0..1
    const int waveN = wave & 3;           // 0..3
    const int blockRow = blockIdx.y * TILE_M;
    const int blockCol = blockIdx.x * TILE_N;

    v8f acc[2][2] = {};

    // staging indices: A tile 64x64 halves, Bt tile 128x64 halves; 16B units
    const int aRow = tid >> 2;            // 0..63
    const int aCol = (tid & 3) * 16;      // 0,16,32,48 -> 2 x b128 (+0, +8)
    const int bRow = tid >> 1;            // 0..127
    const int bCol = (tid & 1) * 32;      // 0,32      -> 4 x b128 (+0..+24)

    const bool aValid = BOUNDED ? ((blockRow + aRow) < M) : true;
    const bool bValid = BOUNDED ? ((blockCol + bRow) < N) : true;
    const _Float16* aBase = A  + (size_t)(aValid ? blockRow + aRow : 0) * K + aCol;
    const _Float16* bBase = Bt + (size_t)(bValid ? blockCol + bRow : 0) * K + bCol;

    auto stageTile = [&](int k0, int buf) {
        stage16(aBase + k0,     &As[buf][aRow][aCol],     aValid);
        stage16(aBase + k0 + 8, &As[buf][aRow][aCol + 8], aValid);
        #pragma unroll
        for (int h = 0; h < 4; ++h)
            stage16(bBase + k0 + h * 8, &Bs[buf][bRow][bCol + h * 8], bValid);
    };

    const int fm  = lane & 15;            // fragment row/col within 16
    const int kg  = lane >> 4;            // K-group select per ISA 7.12.2

    // prologue: stage slab 0, then pipeline: stage t+1 while computing t
    stageTile(0, 0);
    wait_async_stage();
    __syncthreads();

    const int nt = K / TILE_K;
    for (int t = 0; t < nt; ++t) {
        const int buf = t & 1;
        if (t + 1 < nt)
            stageTile((t + 1) * TILE_K, buf ^ 1);

        #pragma unroll
        for (int kc = 0; kc < TILE_K; kc += 32) {
            v16h af[2], bf[2];
            #pragma unroll
            for (int i = 0; i < 2; ++i) {
                const int r = waveM * 32 + i * 16 + fm;
                v8h lo = *reinterpret_cast<const v8h*>(&As[buf][r][kc + kg * 8]);
                v8h hi = *reinterpret_cast<const v8h*>(&As[buf][r][kc + 16 + kg * 8]);
                af[i] = __builtin_shufflevector(lo, hi, 0, 1, 2, 3, 4, 5, 6, 7,
                                                8, 9, 10, 11, 12, 13, 14, 15);
            }
            #pragma unroll
            for (int j = 0; j < 2; ++j) {
                const int c = waveN * 32 + j * 16 + fm;
                v8h lo = *reinterpret_cast<const v8h*>(&Bs[buf][c][kc + kg * 8]);
                v8h hi = *reinterpret_cast<const v8h*>(&Bs[buf][c][kc + 16 + kg * 8]);
                bf[j] = __builtin_shufflevector(lo, hi, 0, 1, 2, 3, 4, 5, 6, 7,
                                                8, 9, 10, 11, 12, 13, 14, 15);
            }
            #pragma unroll
            for (int i = 0; i < 2; ++i)
                #pragma unroll
                for (int j = 0; j < 2; ++j)
                    acc[i][j] = __builtin_amdgcn_wmma_f32_16x16x32_f16(
                        false, af[i], false, bf[j], (short)0, acc[i][j], false, false);
        }

        if (t + 1 < nt) {
            wait_async_stage();           // t+1 slab landed in LDS
            __syncthreads();              // everyone done reading buf, buf^1 ready
        }
    }

    // C/D layout: VGPR r -> row r (lanes 0-15) / row 8+r (lanes 16-31)
    const int ch = lane >> 4;
    #pragma unroll
    for (int i = 0; i < 2; ++i) {
        #pragma unroll
        for (int j = 0; j < 2; ++j) {
            #pragma unroll
            for (int r = 0; r < 8; ++r) {
                const int row = blockRow + waveM * 32 + i * 16 + ch * 8 + r;
                const int col = blockCol + waveN * 32 + j * 16 + fm;
                if (!BOUNDED || (row < M && col < N)) {
                    float v = acc[i][j][r];
                    if (bias) v += bias[row];
                    Cc[(size_t)row * N + col] = v;
                }
            }
        }
    }
}

// ---------------------------------------------------------------------------
// BN batch statistics over (B, N) per channel for a (B, Ch, N) f32 tensor.
// ---------------------------------------------------------------------------
__global__ __launch_bounds__(256)
void bn_stats_kernel(const float* __restrict__ y, float* __restrict__ mean,
                     float* __restrict__ rstd, int Ch, int Nn, int Bb)
{
    __shared__ float s1[256], s2[256];
    const int chn = blockIdx.x;
    const int total = Bb * Nn;
    float sum = 0.f, sq = 0.f;
    for (int idx = threadIdx.x; idx < total; idx += 256) {
        const int b = idx / Nn;
        const int n = idx - b * Nn;
        const float v = y[((size_t)b * Ch + chn) * Nn + n];
        sum += v; sq += v * v;
    }
    s1[threadIdx.x] = sum; s2[threadIdx.x] = sq;
    __syncthreads();
    for (int off = 128; off > 0; off >>= 1) {
        if (threadIdx.x < off) {
            s1[threadIdx.x] += s1[threadIdx.x + off];
            s2[threadIdx.x] += s2[threadIdx.x + off];
        }
        __syncthreads();
    }
    if (threadIdx.x == 0) {
        const float m = s1[0] / (float)total;
        const float var = s2[0] / (float)total - m * m;
        mean[chn] = m;
        rstd[chn] = rsqrtf(var + 1e-5f);
    }
}

// BN + ReLU + cast to f16; transposeOut=1 writes (B, N, Ch) for NT operands.
__global__ void bn_apply_kernel(const float* __restrict__ y,
                                const float* __restrict__ mean, const float* __restrict__ rstd,
                                const float* __restrict__ scale, const float* __restrict__ bias,
                                _Float16* __restrict__ out, int Ch, int Nn, int Bb, int transposeOut)
{
    const size_t total = (size_t)Bb * Ch * Nn;
    for (size_t i = (size_t)blockIdx.x * blockDim.x + threadIdx.x; i < total;
         i += (size_t)gridDim.x * blockDim.x) {
        const size_t n = i % Nn;
        const size_t t = i / Nn;
        const int chn = (int)(t % Ch);
        const int b   = (int)(t / Ch);
        float v = (y[i] - mean[chn]) * rstd[chn] * scale[chn] + bias[chn];
        v = fmaxf(v, 0.f);
        const size_t o = transposeOut ? (((size_t)b * Nn + n) * Ch + chn) : i;
        out[o] = (_Float16)v;
    }
}

__global__ void cast_f32_f16_kernel(const float* __restrict__ in,
                                    _Float16* __restrict__ out, size_t n)
{
    for (size_t i = (size_t)blockIdx.x * blockDim.x + threadIdx.x; i < n;
         i += (size_t)gridDim.x * blockDim.x)
        out[i] = (_Float16)in[i];
}

// (B, Ch, Nn) f32 -> (B, Nn, Ch) f16  (coalesced reads, scattered writes)
__global__ void cast_transpose_kernel(const float* __restrict__ in,
                                      _Float16* __restrict__ out, int Ch, int Nn, int Bb)
{
    const size_t total = (size_t)Bb * Ch * Nn;
    for (size_t i = (size_t)blockIdx.x * blockDim.x + threadIdx.x; i < total;
         i += (size_t)gridDim.x * blockDim.x) {
        const size_t n = i % Nn;
        const size_t t = i / Nn;
        const int chn = (int)(t % Ch);
        const int b   = (int)(t / Ch);
        out[((size_t)b * Nn + n) * Ch + chn] = (_Float16)in[i];
    }
}

// PT[d, n] = rel[(i - x) + L-1, d] with n = x*L + i  (Ph^T, ready as Bt operand)
__global__ void build_PT_kernel(const float* __restrict__ rel, _Float16* __restrict__ PT,
                                int Ldim, int Dk)
{
    const int Nn = Ldim * Ldim;
    const int total = Nn * Dk;
    for (int o = blockIdx.x * blockDim.x + threadIdx.x; o < total;
         o += gridDim.x * blockDim.x) {
        const int n  = o % Nn;
        const int d  = o / Nn;
        const int xr = n / Ldim;
        const int ic = n % Ldim;
        PT[o] = (_Float16)rel[(ic - xr + (Ldim - 1)) * Dk + d];
    }
}

// Row softmax of E (N x N) -> P16 (N x N, f16). One block per row, coalesced.
__global__ __launch_bounds__(256)
void softmax_row_kernel(const float* __restrict__ E, _Float16* __restrict__ P, int Nn)
{
    __shared__ float red[256];
    const float* e = E + (size_t)blockIdx.x * Nn;
    _Float16*    p = P + (size_t)blockIdx.x * Nn;

    float mx = -3.4e38f;
    for (int j = threadIdx.x; j < Nn; j += 256) mx = fmaxf(mx, e[j]);
    red[threadIdx.x] = mx;
    __syncthreads();
    for (int off = 128; off > 0; off >>= 1) {
        if (threadIdx.x < off) red[threadIdx.x] = fmaxf(red[threadIdx.x], red[threadIdx.x + off]);
        __syncthreads();
    }
    mx = red[0];
    __syncthreads();

    float s = 0.f;
    for (int j = threadIdx.x; j < Nn; j += 256) s += __expf(e[j] - mx);
    red[threadIdx.x] = s;
    __syncthreads();
    for (int off = 128; off > 0; off >>= 1) {
        if (threadIdx.x < off) red[threadIdx.x] += red[threadIdx.x + off];
        __syncthreads();
    }
    const float inv = 1.0f / red[0];
    for (int j = threadIdx.x; j < Nn; j += 256)
        p[j] = (_Float16)(__expf(e[j] - mx) * inv);
}

// out = gamma * (attn + BN(EhRaw) + BN(EwRaw)) + x
__global__ void final_combine_kernel(const float* __restrict__ x, const float* __restrict__ attn,
                                     const float* __restrict__ EhR, const float* __restrict__ EwR,
                                     const float* __restrict__ mEh, const float* __restrict__ rEh,
                                     const float* __restrict__ sEh, const float* __restrict__ bEh,
                                     const float* __restrict__ mEw, const float* __restrict__ rEw,
                                     const float* __restrict__ sEw, const float* __restrict__ bEw,
                                     const float* __restrict__ gamma, float* __restrict__ out,
                                     int Ch, int Nn, int Bb)
{
    const size_t total = (size_t)Bb * Ch * Nn;
    const float g = gamma[0];
    for (size_t i = (size_t)blockIdx.x * blockDim.x + threadIdx.x; i < total;
         i += (size_t)gridDim.x * blockDim.x) {
        const int chn = (int)((i / Nn) % Ch);
        const float eh = (EhR[i] - mEh[chn]) * rEh[chn] * sEh[chn] + bEh[chn];
        const float ew = (EwR[i] - mEw[chn]) * rEw[chn] * sEw[chn] + bEw[chn];
        out[i] = g * (attn[i] + eh + ew) + x[i];
    }
}

// ---------------------------------------------------------------------------
// Host orchestration
// ---------------------------------------------------------------------------
static inline size_t alignUp(size_t x) { return (x + 255) & ~(size_t)255; }

extern "C" void kernel_launch(void* const* d_in, const int* in_sizes, int n_in,
                              void* d_out, int out_size, void* d_ws, size_t ws_size,
                              hipStream_t stream)
{
    const float* x        = (const float*)d_in[0];
    const float* Wq       = (const float*)d_in[1];
    const float* bq       = (const float*)d_in[2];
    const float* q_scale  = (const float*)d_in[3];
    const float* q_bias   = (const float*)d_in[4];
    const float* Wk       = (const float*)d_in[5];
    const float* bk       = (const float*)d_in[6];
    const float* k_scale  = (const float*)d_in[7];
    const float* k_bias   = (const float*)d_in[8];
    const float* Wv       = (const float*)d_in[9];
    const float* bv       = (const float*)d_in[10];
    const float* v_scale  = (const float*)d_in[11];
    const float* v_bias   = (const float*)d_in[12];
    const float* rel_rows = (const float*)d_in[13];
    const float* rel_cols = (const float*)d_in[14];
    const float* bnr_s    = (const float*)d_in[15];
    const float* bnr_b    = (const float*)d_in[16];
    const float* bnc_s    = (const float*)d_in[17];
    const float* bnc_b    = (const float*)d_in[18];
    const float* gamma    = (const float*)d_in[19];
    float* out = (float*)d_out;

    char* base = (char*)d_ws;
    size_t off = 0;
    auto carve = [&](size_t bytes) -> char* {
        char* p = base + off;
        off += alignUp(bytes);
        return p;
    };
    _Float16* xT16  = (_Float16*)carve((size_t)kB * kN * kC * 2);    // (B, N, C)
    _Float16* Wq16  = (_Float16*)carve((size_t)kDK * kC * 2);
    _Float16* Wk16  = (_Float16*)carve((size_t)kDK * kC * 2);
    _Float16* Wv16  = (_Float16*)carve((size_t)kC * kC * 2);
    float*    yq    = (float*)   carve((size_t)kB * kDK * kN * 4);
    float*    yk    = (float*)   carve((size_t)kB * kDK * kN * 4);
    float*    yv    = (float*)   carve((size_t)kB * kC  * kN * 4);
    _Float16* qhT   = (_Float16*)carve((size_t)kB * kN * kDK * 2);   // (B, N, DK)
    _Float16* khT   = (_Float16*)carve((size_t)kB * kN * kDK * 2);   // (B, N, DK)
    _Float16* vh    = (_Float16*)carve((size_t)kB * kC  * kN * 2);   // (B, C, N)
    _Float16* PhT16 = (_Float16*)carve((size_t)kDK * kN * 2);        // (DK, N)
    _Float16* PwT16 = (_Float16*)carve((size_t)kDK * kN * 2);
    float*    VPh   = (float*)   carve((size_t)kB * kC * kDK * 4);
    float*    VPw   = (float*)   carve((size_t)kB * kC * kDK * 4);
    _Float16* VPh16 = (_Float16*)carve((size_t)kB * kC * kDK * 2);
    _Float16* VPw16 = (_Float16*)carve((size_t)kB * kC * kDK * 2);
    float*    EhRaw = (float*)   carve((size_t)kB * kC * kN * 4);
    float*    EwRaw = (float*)   carve((size_t)kB * kC * kN * 4);
    float*    attnO = (float*)   carve((size_t)kB * kC * kN * 4);
    float*    Ebuf  = (float*)   carve((size_t)kN * kN * 4);         // per-batch reuse
    _Float16* P16   = (_Float16*)carve((size_t)kN * kN * 2);         // per-batch reuse
    float*    mq  = (float*)carve(kDK * 4);  float* rq  = (float*)carve(kDK * 4);
    float*    mk  = (float*)carve(kDK * 4);  float* rk  = (float*)carve(kDK * 4);
    float*    mv  = (float*)carve(kC * 4);   float* rv  = (float*)carve(kC * 4);
    float*    mEh = (float*)carve(kC * 4);   float* rEh = (float*)carve(kC * 4);
    float*    mEw = (float*)carve(kC * 4);   float* rEw = (float*)carve(kC * 4);
    (void)ws_size; (void)in_sizes; (void)n_in; (void)out_size;

    auto gemm = [&](const _Float16* A, const _Float16* Bt, float* Cm, const float* bias,
                    int M, int N, int K, long long sA, long long sB, long long sC, int batches) {
        dim3 grid((N + TILE_N - 1) / TILE_N, (M + TILE_M - 1) / TILE_M, batches);
        const bool bounded = (M % TILE_M) != 0 || (N % TILE_N) != 0;
        if (bounded)
            wmma_gemm_nt<true><<<grid, 256, 0, stream>>>(A, Bt, Cm, bias, M, N, K, sA, sB, sC);
        else
            wmma_gemm_nt<false><<<grid, 256, 0, stream>>>(A, Bt, Cm, bias, M, N, K, sA, sB, sC);
    };

    // ---- f16 staging (x transposed once: it is the Bt operand of all projections)
    cast_transpose_kernel<<<2048, 256, 0, stream>>>(x, xT16, kC, kN, kB);
    cast_f32_f16_kernel<<<64,  256, 0, stream>>>(Wq, Wq16, (size_t)kDK * kC);
    cast_f32_f16_kernel<<<64,  256, 0, stream>>>(Wk, Wk16, (size_t)kDK * kC);
    cast_f32_f16_kernel<<<512, 256, 0, stream>>>(Wv, Wv16, (size_t)kC * kC);
    build_PT_kernel<<<288, 256, 0, stream>>>(rel_rows, PhT16, kL, kDK);
    build_PT_kernel<<<288, 256, 0, stream>>>(rel_cols, PwT16, kL, kDK);

    // ---- projections: y = W * x   (A = W16 MxC, Bt = xT16 NxC)
    gemm(Wq16, xT16, yq, bq, kDK, kN, kC, 0, (long long)kN * kC, (long long)kDK * kN, kB);
    gemm(Wk16, xT16, yk, bk, kDK, kN, kC, 0, (long long)kN * kC, (long long)kDK * kN, kB);
    gemm(Wv16, xT16, yv, bv, kC,  kN, kC, 0, (long long)kN * kC, (long long)kC  * kN, kB);

    // ---- BN + ReLU + f16 cast (q,k transposed: they are Bt/A operands as (N,DK))
    bn_stats_kernel<<<kDK, 256, 0, stream>>>(yq, mq, rq, kDK, kN, kB);
    bn_stats_kernel<<<kDK, 256, 0, stream>>>(yk, mk, rk, kDK, kN, kB);
    bn_stats_kernel<<<kC,  256, 0, stream>>>(yv, mv, rv, kC,  kN, kB);
    bn_apply_kernel<<<2048, 256, 0, stream>>>(yq, mq, rq, q_scale, q_bias, qhT, kDK, kN, kB, 1);
    bn_apply_kernel<<<2048, 256, 0, stream>>>(yk, mk, rk, k_scale, k_bias, khT, kDK, kN, kB, 1);
    bn_apply_kernel<<<4096, 256, 0, stream>>>(yv, mv, rv, v_scale, v_bias, vh,  kC,  kN, kB, 0);

    // ---- relative-position terms, factored: Eh = (V * Ph) * Q
    gemm(vh, PhT16, VPh, nullptr, kC, kDK, kN, (long long)kC * kN, 0, (long long)kC * kDK, kB);
    gemm(vh, PwT16, VPw, nullptr, kC, kDK, kN, (long long)kC * kN, 0, (long long)kC * kDK, kB);
    cast_f32_f16_kernel<<<1024, 256, 0, stream>>>(VPh, VPh16, (size_t)kB * kC * kDK);
    cast_f32_f16_kernel<<<1024, 256, 0, stream>>>(VPw, VPw16, (size_t)kB * kC * kDK);
    gemm(VPh16, qhT, EhRaw, nullptr, kC, kN, kDK,
         (long long)kC * kDK, (long long)kN * kDK, (long long)kC * kN, kB);
    gemm(VPw16, qhT, EwRaw, nullptr, kC, kN, kDK,
         (long long)kC * kDK, (long long)kN * kDK, (long long)kC * kN, kB);
    bn_stats_kernel<<<kC, 256, 0, stream>>>(EhRaw, mEh, rEh, kC, kN, kB);
    bn_stats_kernel<<<kC, 256, 0, stream>>>(EwRaw, mEw, rEw, kC, kN, kB);

    // ---- attention per batch (sequential on stream; E/P16 buffers reused)
    for (int b = 0; b < kB; ++b) {
        const _Float16* qhTb = qhT + (size_t)b * kN * kDK;
        const _Float16* khTb = khT + (size_t)b * kN * kDK;
        const _Float16* vhb  = vh  + (size_t)b * kC * kN;
        float* attnb = attnO + (size_t)b * kC * kN;
        // E[n][m] = sum_d q[d][n] k[d][m]   (A = qhT, Bt = khT)
        gemm(qhTb, khTb, Ebuf, nullptr, kN, kN, kDK, 0, 0, 0, 1);
        // row softmax -> P16[i][j] (row-major A = attention; exactly the Bt operand)
        softmax_row_kernel<<<kN, 256, 0, stream>>>(Ebuf, P16, kN);
        // out[c][i] = sum_j v[c][j] A[i][j]   (A = vh, Bt = P16)
        gemm(vhb, P16, attnb, nullptr, kC, kN, kN, 0, 0, 0, 1);
    }

    // ---- out = gamma * (attn + BN(Eh) + BN(Ew)) + x
    final_combine_kernel<<<4096, 256, 0, stream>>>(
        x, attnO, EhRaw, EwRaw, mEh, rEh, bnr_s, bnr_b, mEw, rEw, bnc_s, bnc_b,
        gamma, out, kC, kN, kB);
}
